// Attention_53171695125393
// MI455X (gfx1250) — matrix operand order repaired and checked by reading, not compiled, not run on hardware
//
#include <hip/hip_runtime.h>
#include <hip/hip_bf16.h>
#include <stdint.h>

// ---------- types ----------
typedef __bf16 bf16;
typedef __attribute__((ext_vector_type(16))) __bf16 bf16x16;
typedef __attribute__((ext_vector_type(8)))  float  f32x8;
typedef __attribute__((ext_vector_type(4)))  int    v4i_;
typedef __attribute__((address_space(1))) v4i_ as1v4i;   // global int4
typedef __attribute__((address_space(3))) v4i_ as3v4i;   // LDS int4

__device__ __forceinline__ bf16 f2bf(float f) {
    union { float f; uint32_t u; } x; x.f = f;
    uint32_t u = x.u;
    uint32_t r = (u + 0x7FFFu + ((u >> 16) & 1u)) >> 16;
    union { uint16_t s; bf16 b; } y; y.s = (uint16_t)r;
    return y.b;
}

// ---------- async global->LDS (ASYNCcnt path, CDNA5 ISA) ----------
#if defined(__HIP_DEVICE_COMPILE__) && __has_builtin(__builtin_amdgcn_global_load_async_to_lds_b128)
#define HAS_ASYNC 1
#else
#define HAS_ASYNC 0
#endif

__device__ __forceinline__ void cp16(bf16* lds, const bf16* g) {
#if HAS_ASYNC
    v4i_*   gg = (v4i_*)g;        // generic, retype (const stripped)
    v4i_*   ll = (v4i_*)lds;
    as1v4i* g1 = (as1v4i*)gg;     // addrspacecast generic -> global(1)
    as3v4i* l1 = (as3v4i*)ll;     // addrspacecast generic -> lds(3)
    __builtin_amdgcn_global_load_async_to_lds_b128(g1, l1, 0, 0);
#else
    *(uint4*)lds = *(const uint4*)g;
#endif
}

#if HAS_ASYNC
#if __has_builtin(__builtin_amdgcn_s_wait_asynccnt)
#define ASYNC_WAIT(n) __builtin_amdgcn_s_wait_asynccnt(n)
#else
#define ASYNC_WAIT(n) asm volatile("s_wait_asynccnt %0" ::"n"(n) : "memory")
#endif
#else
#define ASYNC_WAIT(n)
#endif

// ---------- sizes ----------
#define BB 4
#define TT 2048
#define CC 1024
#define HH 8
#define DD 128
#define MROWS (BB * TT)          // 8192
#define QKVN  (3 * CC)           // 3072
#define KDIM  1024
#define HEADELEMS ((size_t)TT * DD)
#define BHDELEMS  ((size_t)BB * HH * TT * DD)   // 8388608

// ---------- fp32 -> bf16 convert ----------
__global__ void cvt_f32_bf16(const float* __restrict__ src, bf16* __restrict__ dst, size_t n) {
    size_t i = (size_t)blockIdx.x * blockDim.x + threadIdx.x;
    if (i < n) dst[i] = f2bf(src[i]);
}

// ---------- fp32 [K=1024][N] -> bf16 transposed [N][K=1024] ----------
__global__ void cvt_transpose_bf16(const float* __restrict__ src, bf16* __restrict__ dst,
                                   int N, size_t total) {
    size_t i = (size_t)blockIdx.x * blockDim.x + threadIdx.x;
    if (i >= total) return;
    int n = (int)(i >> 10);
    int k = (int)(i & 1023);
    dst[i] = f2bf(src[(size_t)k * N + n]);
}

// ---------- cross-attention KV routing (heads 5,6 <- p1 ; head 7 <- p0) ----------
__global__ void fill_cross_kv(const float* __restrict__ p0k, const float* __restrict__ p0v,
                              const float* __restrict__ p1k, const float* __restrict__ p1v,
                              bf16* __restrict__ kbf, bf16* __restrict__ vbf,
                              bf16* __restrict__ vtbf) {
    size_t i = (size_t)blockIdx.x * blockDim.x + threadIdx.x;
    const size_t total = (size_t)BB * 3 * TT * DD;
    if (i >= total) return;
    size_t td  = i % HEADELEMS;
    size_t rem = i / HEADELEMS;
    int    hs  = (int)(rem % 3);
    size_t b   = rem / 3;
    int    h   = 5 + hs;
    size_t idx = ((b * HH + h) * HEADELEMS) + td;
    int t = (int)(td >> 7), d = (int)(td & 127);
    const float* sk = (hs < 2) ? p1k : p0k;
    const float* sv = (hs < 2) ? p1v : p0v;
    float vv = sv[idx];
    kbf[idx] = f2bf(sk[idx]);
    vbf[idx] = f2bf(vv);
    vtbf[((b * HH + h) * DD + d) * TT + t] = f2bf(vv);   // [B,H,D,T]
}

// ---------- 64x64 block WMMA GEMM, 4 waves, async double-buffered LDS ----------
// A row-major [M][1024], Bt transposed [N][1024].
// MODE 0: C[m*Nout+n] = acc (fp32)
// MODE 1: QKV epilogue -> q bf16, k/v fp32 outputs + bf16 KV + transposed-V, [B,H,T,D]
template <int MODE>
__global__ __launch_bounds__(128)
void gemm64_wmma(const bf16* __restrict__ A, const bf16* __restrict__ Bt, int Nout,
                 float* __restrict__ Cout,
                 bf16* __restrict__ qbf, bf16* __restrict__ kbf, bf16* __restrict__ vbf,
                 bf16* __restrict__ vtbf, float* __restrict__ kout, float* __restrict__ vout) {
    __shared__ __align__(16) bf16 As[2][64 * 32];   // [row][k]
    __shared__ __align__(16) bf16 Bts[2][64 * 32];  // [n][k]

    const int tid  = threadIdx.x;
    const int wave = tid >> 5;
    const int lane = tid & 31;
    const int hi   = lane >> 4;
    const int ln   = lane & 15;
    const int m0   = blockIdx.x * 64;
    const int n0   = blockIdx.y * 64;

    f32x8 acc[4] = {};

    auto issue_tile = [&](int k0, int buf) {
        #pragma unroll
        for (int c = tid; c < 256; c += 128) {
            int e = c << 3;
            int r = e >> 5, cc = e & 31;
            cp16(&As[buf][e],  A  + (size_t)(m0 + r) * KDIM + k0 + cc);
            cp16(&Bts[buf][e], Bt + (size_t)(n0 + r) * KDIM + k0 + cc);
        }
    };

    issue_tile(0, 0);
    for (int it = 0; it < KDIM / 32; ++it) {
        const int buf = it & 1;
        const bool more = (it + 1) < KDIM / 32;
        if (more) issue_tile((it + 1) * 32, buf ^ 1);
        if (more) { ASYNC_WAIT(4); } else { ASYNC_WAIT(0); }
        __syncthreads();

        // A fragment: 16x32, lane row = 16*wave + ln
        bf16x16 af;
        {
            const bf16* arow = &As[buf][(16 * wave + ln) * 32];
            #pragma unroll
            for (int i = 0; i < 8; ++i) af[i]     = arow[16 * hi + i];
            #pragma unroll
            for (int i = 0; i < 8; ++i) af[8 + i] = arow[16 * hi + 8 + i];
        }
        // prefetch all 4 B fragments, then issue WMMA chain
        bf16x16 bfr[4];
        #pragma unroll
        for (int nt = 0; nt < 4; ++nt) {
            const bf16* brow = &Bts[buf][(16 * nt + ln) * 32 + 16 * hi];
            #pragma unroll
            for (int i = 0; i < 16; ++i) bfr[nt][i] = brow[i];
        }
        #pragma unroll
        for (int nt = 0; nt < 4; ++nt)
            acc[nt] = __builtin_amdgcn_wmma_f32_16x16x32_bf16(
                false, af, false, bfr[nt], (short)0, acc[nt], false, false);
        __syncthreads();
    }

    // epilogue: C layout row m = r + 8*hi, col n = 16*nt + ln
    #pragma unroll
    for (int nt = 0; nt < 4; ++nt) {
        #pragma unroll
        for (int r = 0; r < 8; ++r) {
            int m = m0 + 16 * wave + r + 8 * hi;
            int n = n0 + 16 * nt + ln;
            float v = acc[nt][r];
            if (MODE == 0) {
                Cout[(size_t)m * Nout + n] = v;
            } else {
                int b = m >> 11, t = m & 2047;
                int sec = n >> 10, c = n & 1023;
                int h = c >> 7, d = c & 127;
                size_t idx = ((size_t)(b * HH + h) * TT + t) * DD + d;
                if (sec == 0)      { qbf[idx] = f2bf(v); }
                else if (sec == 1) { kout[idx] = v; kbf[idx] = f2bf(v); }
                else {
                    vout[idx] = v;
                    bf16 bv = f2bf(v);
                    vbf[idx] = bv;
                    vtbf[((size_t)(b * HH + h) * DD + d) * TT + t] = bv;
                }
            }
        }
    }
}

// ---------- flash attention: block = (b,h, 64-query tile), 4 waves x 16 rows ----------
__global__ __launch_bounds__(128)
void flash_attn64(const bf16* __restrict__ qbf, const bf16* __restrict__ kbf,
                  const bf16* __restrict__ vtbf, bf16* __restrict__ att) {
    __shared__ __align__(16) bf16 Ks[2][32 * 128];   // [key][d]
    __shared__ __align__(16) bf16 Vt[2][128 * 32];   // [d][key]
    __shared__ bf16 Pl[4][16][32];                   // per-wave P scratch

    const int tid  = threadIdx.x;
    const int wave = tid >> 5;
    const int lane = tid & 31;
    const int hi   = lane >> 4;
    const int ln   = lane & 15;

    const int bh    = blockIdx.x;       // 0..31
    const int qt    = blockIdx.y;       // 0..31
    const int qbase = qt * 64;
    const size_t base  = (size_t)bh * TT * DD;   // [B,H,T,D] head base
    const size_t baseT = (size_t)bh * DD * TT;   // [B,H,D,T] head base

    // Q fragments (16 rows x 128 d) resident in registers
    bf16x16 qf[4];
    {
        const bf16* qr = qbf + base + (size_t)(qbase + 16 * wave + ln) * DD;
        #pragma unroll
        for (int dk = 0; dk < 4; ++dk) {
            #pragma unroll
            for (int i = 0; i < 8; ++i) qf[dk][i]     = qr[32 * dk + 16 * hi + i];
            #pragma unroll
            for (int i = 0; i < 8; ++i) qf[dk][8 + i] = qr[32 * dk + 16 * hi + 8 + i];
        }
    }

    f32x8 O[8] = {};
    float ms[8], ls[8];
    #pragma unroll
    for (int r = 0; r < 8; ++r) { ms[r] = -1e30f; ls[r] = 0.f; }
    const float scale = 0.08838834764831845f;  // 1/sqrt(128)

    auto issue_kv = [&](int kb, int buf) {
        #pragma unroll
        for (int c = tid; c < 512; c += 128) {
            int e = c << 3;
            int key = e >> 7, d = e & 127;          // K tile [32][128]
            cp16(&Ks[buf][e], kbf + base + (size_t)(kb + key) * DD + d);
            int dr = e >> 5, kk = e & 31;           // Vt tile [128][32]
            cp16(&Vt[buf][e], vtbf + baseT + (size_t)dr * TT + kb + kk);
        }
    };

    const int ntiles = 2 * qt + 2;  // causal: keys up to qbase+63
    issue_kv(0, 0);
    for (int j = 0; j < ntiles; ++j) {
        const int kb = j * 32;
        const int buf = j & 1;
        const bool more = (j + 1) < ntiles;
        if (more) issue_kv(kb + 32, buf ^ 1);
        if (more) { ASYNC_WAIT(8); } else { ASYNC_WAIT(0); }
        __syncthreads();

        // S = Q K^T : prefetch all 8 B-fragments, then 8 WMMAs
        bf16x16 bk[8];
        #pragma unroll
        for (int dk = 0; dk < 4; ++dk) {
            const bf16* kr0 = &Ks[buf][(size_t)ln * DD + 32 * dk + 16 * hi];
            const bf16* kr1 = &Ks[buf][(size_t)(16 + ln) * DD + 32 * dk + 16 * hi];
            #pragma unroll
            for (int i = 0; i < 16; ++i) bk[2 * dk][i]     = kr0[i];
            #pragma unroll
            for (int i = 0; i < 16; ++i) bk[2 * dk + 1][i] = kr1[i];
        }
        f32x8 S0 = {}, S1 = {};
        #pragma unroll
        for (int dk = 0; dk < 4; ++dk) {
            S0 = __builtin_amdgcn_wmma_f32_16x16x32_bf16(false, qf[dk], false, bk[2 * dk],     (short)0, S0, false, false);
            S1 = __builtin_amdgcn_wmma_f32_16x16x32_bf16(false, qf[dk], false, bk[2 * dk + 1], (short)0, S1, false, false);
        }

        // online softmax (rows r+8*hi per lane; 16-lane row reductions)
        float alpha[8], p0v[8], p1v[8];
        #pragma unroll
        for (int r = 0; r < 8; ++r) {
            int qrow = qbase + 16 * wave + r + 8 * hi;
            int k0i = kb + ln, k1i = kb + 16 + ln;
            bool ok0 = (k0i <= qrow), ok1 = (k1i <= qrow);
            float s0 = ok0 ? S0[r] * scale : -1e30f;
            float s1 = ok1 ? S1[r] * scale : -1e30f;
            float mx = fmaxf(s0, s1);
            #pragma unroll
            for (int off = 1; off < 16; off <<= 1) mx = fmaxf(mx, __shfl_xor(mx, off, 32));
            float mnew = fmaxf(ms[r], mx);
            float a = __expf(ms[r] - mnew);
            float p0 = ok0 ? __expf(s0 - mnew) : 0.f;
            float p1 = ok1 ? __expf(s1 - mnew) : 0.f;
            float rs = p0 + p1;
            #pragma unroll
            for (int off = 1; off < 16; off <<= 1) rs += __shfl_xor(rs, off, 32);
            ls[r] = ls[r] * a + rs;
            ms[r] = mnew;
            alpha[r] = a; p0v[r] = p0; p1v[r] = p1;
        }
        #pragma unroll
        for (int ot = 0; ot < 8; ++ot)
            #pragma unroll
            for (int r = 0; r < 8; ++r) O[ot][r] *= alpha[r];

        // P: C layout -> LDS -> A layout (same-wave round trip)
        #pragma unroll
        for (int r = 0; r < 8; ++r) {
            int m = r + 8 * hi;
            Pl[wave][m][ln]      = f2bf(p0v[r]);
            Pl[wave][m][16 + ln] = f2bf(p1v[r]);
        }
        bf16x16 pf;
        {
            const bf16* pr = &Pl[wave][ln][0];
            #pragma unroll
            for (int i = 0; i < 8; ++i) pf[i]     = pr[16 * hi + i];
            #pragma unroll
            for (int i = 0; i < 8; ++i) pf[8 + i] = pr[16 * hi + 8 + i];
        }

        // O += P x V : prefetch all 8 V-fragments, then 8 WMMAs
        bf16x16 vfr[8];
        #pragma unroll
        for (int ot = 0; ot < 8; ++ot) {
            const bf16* vr = &Vt[buf][(16 * ot + ln) * 32 + 16 * hi];
            #pragma unroll
            for (int i = 0; i < 16; ++i) vfr[ot][i] = vr[i];
        }
        #pragma unroll
        for (int ot = 0; ot < 8; ++ot)
            O[ot] = __builtin_amdgcn_wmma_f32_16x16x32_bf16(
                false, pf, false, vfr[ot], (short)0, O[ot], false, false);
        __syncthreads();
    }

    // normalize and write att[B,T,C] as bf16 (feeds proj GEMM)
    const size_t b = (size_t)(bh >> 3);
    const int h = bh & 7;
    #pragma unroll
    for (int ot = 0; ot < 8; ++ot) {
        #pragma unroll
        for (int r = 0; r < 8; ++r) {
            int qrow = qbase + 16 * wave + r + 8 * hi;
            float inv = 1.0f / ls[r];
            att[((b * TT + qrow) * CC) + (size_t)h * DD + 16 * ot + ln] = f2bf(O[ot][r] * inv);
        }
    }
}

extern "C" void kernel_launch(void* const* d_in, const int* in_sizes, int n_in,
                              void* d_out, int out_size, void* d_ws, size_t ws_size,
                              hipStream_t stream) {
    const float* x     = (const float*)d_in[0];
    const float* Wattn = (const float*)d_in[1];
    const float* Wproj = (const float*)d_in[2];
    const float* p0k   = (const float*)d_in[3];
    const float* p0v   = (const float*)d_in[4];
    const float* p1k   = (const float*)d_in[5];
    const float* p1v   = (const float*)d_in[6];

    float* y    = (float*)d_out;                 // [B,T,C]
    float* kout = y + (size_t)MROWS * CC;        // [B,H,T,D]
    float* vout = kout + BHDELEMS;               // [B,H,T,D]

    char* w = (char*)d_ws;
    bf16* xbf   = (bf16*)w;  w += (size_t)MROWS * CC * 2;    // 16.8 MB
    bf16* wabt  = (bf16*)w;  w += (size_t)CC * QKVN * 2;     //  6.3 MB  (transposed [3072][1024])
    bf16* wpbt  = (bf16*)w;  w += (size_t)CC * CC * 2;       //  2.1 MB  (transposed [1024][1024])
    bf16* qbf   = (bf16*)w;  w += BHDELEMS * 2;              // 16.8 MB
    bf16* kbf   = (bf16*)w;  w += BHDELEMS * 2;              // 16.8 MB
    bf16* vbf   = (bf16*)w;  w += BHDELEMS * 2;              // 16.8 MB
    bf16* vtbf  = (bf16*)w;  w += BHDELEMS * 2;              // 16.8 MB  ([B,H,D,T])
    bf16* attbf = (bf16*)w;                                  // 16.8 MB   (total ~109 MB)

    // 1. bf16 conversions (weights transposed for K-contiguous async staging)
    {
        size_t n1 = (size_t)MROWS * CC;
        cvt_f32_bf16<<<(unsigned)((n1 + 255) / 256), 256, 0, stream>>>(x, xbf, n1);
        size_t n2 = (size_t)CC * QKVN;
        cvt_transpose_bf16<<<(unsigned)((n2 + 255) / 256), 256, 0, stream>>>(Wattn, wabt, QKVN, n2);
        size_t n3 = (size_t)CC * CC;
        cvt_transpose_bf16<<<(unsigned)((n3 + 255) / 256), 256, 0, stream>>>(Wproj, wpbt, CC, n3);
    }
    // 2. QKV GEMM with scatter epilogue (k/v fp32 outputs written here)
    gemm64_wmma<1><<<dim3(MROWS / 64, QKVN / 64), 128, 0, stream>>>(
        xbf, wabt, QKVN, nullptr, qbf, kbf, vbf, vtbf, kout, vout);
    // 3. route cross-attention KV (heads 5-7)
    {
        size_t n = (size_t)BB * 3 * TT * DD;
        fill_cross_kv<<<(unsigned)((n + 255) / 256), 256, 0, stream>>>(
            p0k, p0v, p1k, p1v, kbf, vbf, vtbf);
    }
    // 4. flash attention
    flash_attn64<<<dim3(BB * HH, TT / 64), 128, 0, stream>>>(qbf, kbf, vtbf, attbf);
    // 5. output projection
    gemm64_wmma<0><<<dim3(MROWS / 64, CC / 64), 128, 0, stream>>>(
        attbf, wpbt, CC, y, nullptr, nullptr, nullptr, nullptr, nullptr, nullptr);
}
